// DCBasicBlock_70712341562031
// MI455X (gfx1250) — 16-bit and 8-bit matrix operand order repaired and checked by reading, compiled, not run on hardware
//
#include <hip/hip_runtime.h>
#include <hip/hip_bf16.h>

typedef __attribute__((ext_vector_type(16))) _Float16 v16h;
typedef __attribute__((ext_vector_type(8)))  _Float16 v8h;
typedef __attribute__((ext_vector_type(8)))  float    v8f;
typedef __attribute__((ext_vector_type(2)))  int      v2i;
typedef __attribute__((ext_vector_type(4)))  int      v4i;
typedef __attribute__((ext_vector_type(8)))  int      v8i;
typedef __attribute__((ext_vector_type(16))) int      v16i;

#define DEV __device__ __forceinline__

constexpr int Nn = 32, Cc = 256, Hh = 56, Ww = 56;
constexpr int XW  = 72;
constexpr int Y1H = 54;
constexpr int YW  = 64;
constexpr int OH  = 48, OW = 56;

constexpr size_t SZ_X8  = (size_t)Nn * Hh * XW * Cc;
constexpr size_t SZ_Y1  = (size_t)Nn * Y1H * YW * Cc * 4;
constexpr size_t SZ_WB1 = (size_t)9 * 256 * 256;
constexpr size_t OFF_X8  = 0;
constexpr size_t OFF_Y1  = OFF_X8  + SZ_X8;
constexpr size_t OFF_WB1 = OFF_Y1  + SZ_Y1;
constexpr size_t OFF_WB2 = OFF_WB1 + SZ_WB1;

__device__ __constant__ unsigned char E4M3_TAB[8] =
    {0x00, 0x38, 0x40, 0x44, 0x48, 0x4A, 0x4C, 0x4E};

DEV unsigned ldsAddr(const void* p) {
  return (unsigned)(uintptr_t)p;
}
DEV void asyncLdsB128(unsigned ldsOff, unsigned vOff, const void* sBase) {
  asm volatile("global_load_async_to_lds_b128 %0, %1, %2"
               :: "v"(ldsOff), "v"(vOff), "s"(sBase)
               : "memory");
}
DEV void waitAsync0() { asm volatile("s_wait_asynccnt 0" ::: "memory"); }

DEV unsigned pack2(_Float16 a, _Float16 b) {
  unsigned short ua = __builtin_bit_cast(unsigned short, a);
  unsigned short ub = __builtin_bit_cast(unsigned short, b);
  return (unsigned)ua | ((unsigned)ub << 16);
}

DEV v8f wmma_f16(v16h a, v16h b, v8f c) {
  return __builtin_amdgcn_wmma_f32_16x16x32_f16(
      false, a, false, b, (short)0, c, false, false);
}
DEV v8f wmma_fp8(v16i a, v16i b, v8f c) {
  return __builtin_amdgcn_wmma_f32_16x16x128_fp8_fp8(
      a, b, (short)0, c, false, false);
}

DEV v16h fragA(const _Float16* tile, int row, int lane) {
  const _Float16* p = tile + row * 32 + ((lane & 16) ? 8 : 0);
  v8h a0 = *(const v8h*)(p);
  v8h a1 = *(const v8h*)(p + 16);
  return __builtin_shufflevector(a0, a1,
      0, 1, 2, 3, 4, 5, 6, 7, 8, 9, 10, 11, 12, 13, 14, 15);
}
DEV v16h fragB(const _Float16* tile, int row, int lane) {
  const _Float16* p = tile + row * 32 + ((lane & 16) ? 8 : 0);
  v8h b0 = *(const v8h*)(p);
  v8h b1 = *(const v8h*)(p + 16);
  return __builtin_shufflevector(b0, b1,
      0, 1, 2, 3, 4, 5, 6, 7, 8, 9, 10, 11, 12, 13, 14, 15);
}

DEV v16i fragA8(const char* rowBase, int lane) {
  const char* p = rowBase + ((lane & 16) ? 16 : 0);
  v2i q0 = *(const v2i*)(p);
  v2i q1 = *(const v2i*)(p + 8);
  v2i q2 = *(const v2i*)(p + 32);
  v2i q3 = *(const v2i*)(p + 40);
  v2i q4 = *(const v2i*)(p + 64);
  v2i q5 = *(const v2i*)(p + 72);
  v2i q6 = *(const v2i*)(p + 96);
  v2i q7 = *(const v2i*)(p + 104);
  v4i a = __builtin_shufflevector(q0, q1, 0, 1, 2, 3);
  v4i b = __builtin_shufflevector(q2, q3, 0, 1, 2, 3);
  v4i c = __builtin_shufflevector(q4, q5, 0, 1, 2, 3);
  v4i d = __builtin_shufflevector(q6, q7, 0, 1, 2, 3);
  v8i ab = __builtin_shufflevector(a, b, 0, 1, 2, 3, 4, 5, 6, 7);
  v8i cd = __builtin_shufflevector(c, d, 0, 1, 2, 3, 4, 5, 6, 7);
  return __builtin_shufflevector(ab, cd,
      0, 1, 2, 3, 4, 5, 6, 7, 8, 9, 10, 11, 12, 13, 14, 15);
}
DEV v16i fragB8(const char* rowBase, int lane) {
  const char* p = rowBase + ((lane & 16) ? 16 : 0);
  v4i a = *(const v4i*)(p);
  v4i b = *(const v4i*)(p + 32);
  v4i c = *(const v4i*)(p + 64);
  v4i d = *(const v4i*)(p + 96);
  v8i ab = __builtin_shufflevector(a, b, 0, 1, 2, 3, 4, 5, 6, 7);
  v8i cd = __builtin_shufflevector(c, d, 0, 1, 2, 3, 4, 5, 6, 7);
  return __builtin_shufflevector(ab, cd,
      0, 1, 2, 3, 4, 5, 6, 7, 8, 9, 10, 11, 12, 13, 14, 15);
}

__global__ void k_pack_x8(const float* __restrict__ x, unsigned char* __restrict__ X8) {
  size_t e = (size_t)blockIdx.x * blockDim.x + threadIdx.x;
  constexpr size_t total = (size_t)Nn * Hh * XW * Cc;
  if (e >= total) return;
  int ci = (int)(e & 255);
  size_t r = e >> 8;
  int wp = (int)(r % XW);
  size_t r2 = r / XW;
  int h = (int)(r2 % Hh);
  int n = (int)(r2 / Hh);
  unsigned char b = 0;
  if (wp < Ww) {
    int v = (int)x[((size_t)(n * Cc + ci) * Hh + h) * Ww + wp];
    b = E4M3_TAB[v & 7];
  }
  X8[e] = b;
}
__global__ void k_pack_w1_8(const float* __restrict__ w, unsigned char* __restrict__ WB) {
  int e = blockIdx.x * blockDim.x + threadIdx.x;
  if (e >= 9 * 256 * 256) return;
  int pos = e >> 16, rem = e & 65535;
  int co = rem >> 8, ci = rem & 255;
  int v = (int)w[(co * 256 + ci) * 9 + pos];
  WB[e] = E4M3_TAB[v & 7];
}
__global__ void k_pack_w2(const float* __restrict__ w, _Float16* __restrict__ WB) {
  int e = blockIdx.x * blockDim.x + threadIdx.x;
  if (e >= 9 * 256 * 512) return;
  int pos = e / (256 * 512);
  int rem = e % (256 * 512);
  int co = rem >> 9, k2 = rem & 511, ci = k2 >> 1;
  WB[e] = (_Float16)w[(co * 256 + ci) * 9 + pos];
}

__global__ __launch_bounds__(256)
void k_conv1(const unsigned char* __restrict__ X8,
             const unsigned char* __restrict__ WB1,
             unsigned* __restrict__ Y1u) {
  __shared__ alignas(128) char shX[2][64 * 128];
  __shared__ alignas(128) char shW[2][64 * 128];

  const int t = threadIdx.x;
  const int lane = t & 31;
  const int wid  = t >> 5;
  const int waveW  = (wid >> 1) << 4;
  const int waveCo = (wid & 1) << 5;

  const int co0 = blockIdx.x << 6;
  const int n = blockIdx.y / Y1H;
  const int h = blockIdx.y % Y1H;

  unsigned vOffc[2], ldsRelc[2];
#pragma unroll
  for (int i = 0; i < 2; ++i) {
    const int id = t + (i << 8);
    vOffc[i]   = ((unsigned)(id >> 3) << 8) | ((unsigned)(id & 7) << 4);
    ldsRelc[i] = ((unsigned)(id >> 3) << 7) | ((unsigned)(id & 7) << 4);
  }
  unsigned ldsX[2][2], ldsW[2][2];
#pragma unroll
  for (int bfr = 0; bfr < 2; ++bfr)
#pragma unroll
    for (int i = 0; i < 2; ++i) {
      ldsX[bfr][i] = ldsAddr(&shX[bfr][0]) + ldsRelc[i];
      ldsW[bfr][i] = ldsAddr(&shW[bfr][0]) + ldsRelc[i];
    }

  auto issue = [&](int s, int buf) {
    const int pos = s >> 1;
    const int ci0 = (s & 1) << 7;
    const int kh = pos / 3, kw = pos - kh * 3;
    const char* bX = (const char*)X8 +
        (((size_t)(n * Hh + h + kh) * XW + kw) * 256 + ci0);
    const char* bW = (const char*)WB1 +
        (((size_t)(pos * 256 + co0)) * 256 + ci0);
#pragma unroll
    for (int i = 0; i < 2; ++i) {
      asyncLdsB128(ldsX[buf][i], vOffc[i], bX);
      asyncLdsB128(ldsW[buf][i], vOffc[i], bW);
    }
  };

  v8f acc0 = {}, acc1 = {};
  constexpr int STEPS = 9 * 2;

  issue(0, 0);
  waitAsync0();
  __syncthreads();

  for (int s = 0; s < STEPS; ++s) {
    const int cur = s & 1;
    if (s + 1 < STEPS) issue(s + 1, cur ^ 1);

    v16i a  = fragA8(shX[cur] + (waveW + (lane & 15)) * 128, lane);
    v16i b0 = fragB8(shW[cur] + (waveCo + (lane & 15)) * 128, lane);
    v16i b1 = fragB8(shW[cur] + (waveCo + 16 + (lane & 15)) * 128, lane);
    acc0 = wmma_fp8(a, b0, acc0);
    acc1 = wmma_fp8(a, b1, acc1);

    waitAsync0();
    __syncthreads();
  }

  const int nl = lane & 15;
  const int mh = (lane >> 4) << 3;
#pragma unroll
  for (int j = 0; j < 2; ++j) {
    const v8f a = j ? acc1 : acc0;
    const int co = co0 + waveCo + (j << 4) + nl;
#pragma unroll
    for (int r = 0; r < 8; ++r) {
      const int w = waveW + r + mh;
      const float v = a[r];
      const _Float16 hi = (_Float16)v;
      const _Float16 lo = (_Float16)(v - (float)hi);
      Y1u[((size_t)(n * Y1H + h) * YW + w) * 256 + co] = pack2(hi, lo);
    }
  }
}

__global__ __launch_bounds__(256)
void k_conv2(const unsigned* __restrict__ Y1u,
             const _Float16* __restrict__ WB2,
             const float* __restrict__ xin,
             float* __restrict__ out) {
  __shared__ alignas(128) _Float16 shW[2][64 * 32];
  __shared__ alignas(128) unsigned shXu[2][64 * 16];

  const int t = threadIdx.x;
  const int lane = t & 31;
  const int wid  = t >> 5;
  const int waveCo = (wid >> 1) << 4;
  const int waveW  = (wid & 1) << 5;

  const int co0 = blockIdx.x << 6;
  const int n = blockIdx.y / OH;
  const int h = blockIdx.y % OH;

  const unsigned vOff   = ((unsigned)(t >> 2) << 10) | ((unsigned)(t & 3) << 4);
  const unsigned ldsRel = ((unsigned)(t >> 2) << 6)  | ((unsigned)(t & 3) << 4);
  const unsigned ldsX[2] = { ldsAddr(&shXu[0][0]) + ldsRel, ldsAddr(&shXu[1][0]) + ldsRel };
  const unsigned ldsW[2] = { ldsAddr(&shW[0][0])  + ldsRel, ldsAddr(&shW[1][0])  + ldsRel };

  auto issue = [&](int s, int buf) {
    const int pos = s >> 4;
    const int k0  = (s & 15) << 5;
    const int kh = pos / 3, kw = pos - kh * 3;
    const int cib = k0 >> 1;
    const char* bX = (const char*)Y1u +
        ((((size_t)(n * Y1H + h + 2 + kh) * YW + 1 + kw) * 256 + cib) << 2);
    const char* bW = (const char*)WB2 +
        ((((size_t)(pos * 256 + co0)) * 512 + k0) << 1);
    asyncLdsB128(ldsX[buf], vOff, bX);
    asyncLdsB128(ldsW[buf], vOff, bW);
  };

  v8f acc0 = {}, acc1 = {};
  constexpr int STEPS = 9 * 16;

  issue(0, 0);
  waitAsync0();
  __syncthreads();

  for (int s = 0; s < STEPS; ++s) {
    const int cur = s & 1;
    if (s + 1 < STEPS) issue(s + 1, cur ^ 1);

    const _Float16* xt = (const _Float16*)shXu[cur];
    v16h a  = fragA(shW[cur], waveCo + (lane & 15), lane);
    v16h b0 = fragB(xt, waveW + (lane & 15), lane);
    v16h b1 = fragB(xt, waveW + 16 + (lane & 15), lane);
    acc0 = wmma_f16(a, b0, acc0);
    acc1 = wmma_f16(a, b1, acc1);

    waitAsync0();
    __syncthreads();
  }

  const int nl = lane & 15;
  const int mh = (lane >> 4) << 3;
#pragma unroll
  for (int j = 0; j < 2; ++j) {
    const v8f a = j ? acc1 : acc0;
    const int w = waveW + (j << 4) + nl;
    if (w < OW) {
#pragma unroll
      for (int r = 0; r < 8; ++r) {
        const int co = co0 + waveCo + r + mh;
        const float v =
            a[r] + xin[((size_t)(n * Cc + co) * Hh + (h + 4)) * Ww + w];
        out[((size_t)(n * Cc + co) * OH + h) * OW + w] = fmaxf(v, 0.0f);
      }
    }
  }
}

extern "C" void kernel_launch(void* const* d_in, const int* in_sizes, int n_in,
                              void* d_out, int out_size, void* d_ws, size_t ws_size,
                              hipStream_t stream) {
  (void)in_sizes; (void)n_in; (void)out_size; (void)ws_size;
  const float* x  = (const float*)d_in[0];
  const float* w1 = (const float*)d_in[1];
  const float* w2 = (const float*)d_in[2];
  float* out = (float*)d_out;
  char* ws = (char*)d_ws;

  unsigned char* X8  = (unsigned char*)(ws + OFF_X8);
  unsigned*      Y1u = (unsigned*)(ws + OFF_Y1);
  unsigned char* WB1 = (unsigned char*)(ws + OFF_WB1);
  _Float16*      WB2 = (_Float16*)(ws + OFF_WB2);

  {
    const size_t tot = (size_t)Nn * Hh * XW * Cc;
    k_pack_x8<<<(unsigned)((tot + 255) / 256), 256, 0, stream>>>(x, X8);
  }
  k_pack_w1_8<<<(9 * 256 * 256 + 255) / 256, 256, 0, stream>>>(w1, WB1);
  k_pack_w2<<<(9 * 256 * 512 + 255) / 256, 256, 0, stream>>>(w2, WB2);

  k_conv1<<<dim3(4, Nn * Y1H), 256, 0, stream>>>(X8, WB1, Y1u);
  k_conv2<<<dim3(4, Nn * OH), 256, 0, stream>>>(Y1u, WB2, x, out);
}
